// SelfAttention_5909874999592
// MI455X (gfx1250) — matrix operand order repaired and checked by reading, not compiled, not run on hardware
//
#include <hip/hip_runtime.h>
#include <math.h>

// Problem constants (match reference)
#define B_   8
#define S_   2048
#define DIN  1024
#define DH   1024

typedef __bf16 bf16_t;
typedef __attribute__((ext_vector_type(16))) __bf16 v16bf;
typedef __attribute__((ext_vector_type(8)))  __bf16 v8bf;
typedef __attribute__((ext_vector_type(4)))  __bf16 v4bf;
typedef __attribute__((ext_vector_type(8)))  float  v8f;

union Frag16 { v16bf v; v8bf h[2]; };

static __device__ __forceinline__ v8f wmma_bf16(const Frag16& a, const Frag16& b, v8f c) {
  // D = A(16x32 bf16) * B(32x16 bf16) + C(16x16 f32)
  return __builtin_amdgcn_wmma_f32_16x16x32_bf16(false, a.v, false, b.v,
                                                 (short)0, c, false, false);
}

// ---------------------------------------------------------------------------
// 1) fp32 -> bf16 conversion of x (vectorized float4 -> 4x bf16)
// ---------------------------------------------------------------------------
__global__ void k_cvt_x(const float* __restrict__ x, bf16_t* __restrict__ xb, int n4) {
  int i = blockIdx.x * blockDim.x + threadIdx.x;
  if (i >= n4) return;
  float4 v = ((const float4*)x)[i];
  v4bf o;
  o[0] = (__bf16)v.x; o[1] = (__bf16)v.y; o[2] = (__bf16)v.z; o[3] = (__bf16)v.w;
  ((v4bf*)xb)[i] = o;
}

// ---------------------------------------------------------------------------
// 2) W[k][n] (fp32) -> WT[n][k] (bf16), blockIdx.y selects q/k/v
// ---------------------------------------------------------------------------
__global__ void k_cvt_w(const float* __restrict__ Wq, const float* __restrict__ Wk,
                        const float* __restrict__ Wv,
                        bf16_t* __restrict__ Tq, bf16_t* __restrict__ Tk,
                        bf16_t* __restrict__ Tv) {
  int idx = blockIdx.x * blockDim.x + threadIdx.x;   // 0 .. DIN*DH-1
  const float* W = (blockIdx.y == 0) ? Wq : ((blockIdx.y == 1) ? Wk : Wv);
  bf16_t*      T = (blockIdx.y == 0) ? Tq : ((blockIdx.y == 1) ? Tk : Tv);
  int n = idx & (DH - 1);
  int k = idx >> 10;
  T[(size_t)n * DIN + k] = (__bf16)W[(size_t)k * DH + n];  // coalesced read, scattered write
}

// ---------------------------------------------------------------------------
// 3) QKV projection GEMM: C[16384,1024] = Xb @ W + bias  (bf16 in, bf16 out)
//    Block tile 128x128, 8 waves (4 M x 2 N), wave tile 32x64 (2x4 WMMA frags).
//    transposeV: store output as VT[b][d][s] instead of row-major.
// ---------------------------------------------------------------------------
__global__ __launch_bounds__(256) void k_gemm_qkv(
    const bf16_t* __restrict__ Xb, const bf16_t* __restrict__ WT,
    const float* __restrict__ bias, bf16_t* __restrict__ Out, int transposeV) {
  __shared__ __attribute__((aligned(16))) bf16_t xs[128 * 40];  // 128 rows, pad->40
  __shared__ __attribute__((aligned(16))) bf16_t wsd[128 * 40];
  const int tid  = threadIdx.x;
  const int wave = tid >> 5;
  const int lane = tid & 31;
  const int half = lane >> 4;
  const int ln   = lane & 15;
  const int wm   = wave >> 1;   // 0..3
  const int wn   = wave & 1;    // 0..1
  const int mb   = blockIdx.y;  // 0..127
  const int nb   = blockIdx.x;  // 0..7

  v8f acc[2][4];
  v8f zero = {};
  for (int mi = 0; mi < 2; ++mi)
    for (int ni = 0; ni < 4; ++ni) acc[mi][ni] = zero;

  const int lr = tid >> 1;          // staging row 0..127
  const int lc = (tid & 1) * 16;    // staging col 0 / 16

  for (int k0 = 0; k0 < DIN; k0 += 32) {
    {
      const v8bf* sx = (const v8bf*)(Xb + (size_t)(mb * 128 + lr) * DIN + k0 + lc);
      v8bf a0 = sx[0], a1 = sx[1];
      *(v8bf*)(xs + lr * 40 + lc)     = a0;
      *(v8bf*)(xs + lr * 40 + lc + 8) = a1;
      const v8bf* sw = (const v8bf*)(WT + (size_t)(nb * 128 + lr) * DIN + k0 + lc);
      v8bf b0 = sw[0], b1 = sw[1];
      *(v8bf*)(wsd + lr * 40 + lc)     = b0;
      *(v8bf*)(wsd + lr * 40 + lc + 8) = b1;
    }
    __syncthreads();
    Frag16 a[2], b[4];
    for (int mi = 0; mi < 2; ++mi) {
      const bf16_t* p = xs + (wm * 32 + mi * 16 + ln) * 40 + half * 8;
      a[mi].h[0] = *(const v8bf*)(p);
      a[mi].h[1] = *(const v8bf*)(p + 16);
    }
    for (int ni = 0; ni < 4; ++ni) {
      const bf16_t* p = wsd + (wn * 64 + ni * 16 + ln) * 40 + half * 8;
      b[ni].h[0] = *(const v8bf*)(p);
      b[ni].h[1] = *(const v8bf*)(p + 16);
    }
    for (int mi = 0; mi < 2; ++mi)
      for (int ni = 0; ni < 4; ++ni)
        acc[mi][ni] = wmma_bf16(a[mi], b[ni], acc[mi][ni]);
    __syncthreads();
  }

  // Epilogue: bias add, cast to bf16, store (optionally transposed for V)
  for (int ni = 0; ni < 4; ++ni) {
    const int cg = nb * 128 + wn * 64 + ni * 16 + ln;   // global column (d)
    const float bv = bias[cg];
    for (int mi = 0; mi < 2; ++mi) {
      const int rbase = mb * 128 + wm * 32 + mi * 16 + half * 8;
      v8f r = acc[mi][ni];
      if (!transposeV) {
        for (int e = 0; e < 8; ++e)
          Out[(size_t)(rbase + e) * DH + cg] = (__bf16)(r[e] + bv);
      } else {
        for (int e = 0; e < 8; ++e) {
          int row = rbase + e;
          int bb  = row >> 11;       // batch
          int s   = row & 2047;      // seq pos
          Out[((size_t)bb * DH + cg) * S_ + s] = (__bf16)(r[e] + bv);
        }
      }
    }
  }
}

// ---------------------------------------------------------------------------
// 4) Flash attention: one block = (batch, 16 queries). 8 waves.
//    Score phase: wave w handles keys [kc*128 + w*16, +16) over full D.
//    PV phase:    wave w handles output columns [w*128, +128).
// ---------------------------------------------------------------------------
__global__ __launch_bounds__(256) void k_attn(
    const bf16_t* __restrict__ Qb, const bf16_t* __restrict__ Kb,
    const bf16_t* __restrict__ VT, float* __restrict__ Out) {
  __shared__ __attribute__((aligned(16))) bf16_t qs[16 * 1032];  // Q tile, padded
  __shared__ __attribute__((aligned(16))) bf16_t ps[16 * 136];   // P tile, padded
  __shared__ float red_m[8][16];
  __shared__ float red_s[8][16];

  const int tid  = threadIdx.x;
  const int w    = tid >> 5;
  const int lane = tid & 31;
  const int half = lane >> 4;
  const int ln   = lane & 15;
  const int q0   = blockIdx.x * 16;
  const int b    = blockIdx.y;

  // Stage Q tile (16 x 1024 bf16) into LDS
  {
    const int row = tid >> 4;            // 0..15
    const int cb  = (tid & 15) * 64;     // 64 cols per thread
    const bf16_t* src = Qb + ((size_t)(b * S_ + q0 + row)) * DH + cb;
    bf16_t* dst = qs + row * 1032 + cb;
    for (int j = 0; j < 8; ++j)
      *(v8bf*)(dst + j * 8) = *(const v8bf*)(src + j * 8);
  }
  __syncthreads();

  v8f o[8];
  v8f zero = {};
  for (int ni = 0; ni < 8; ++ni) o[ni] = zero;
  float m_run[8], l_run[8];
  for (int r = 0; r < 8; ++r) { m_run[r] = -1e30f; l_run[r] = 0.0f; }
  const float scale = 0.03125f;  // 1/sqrt(1024)

  for (int kc = 0; kc < S_ / 128; ++kc) {
    const int kb = kc * 128 + w * 16;

    // ---- scores: S(16q x 16k) = Q @ K^T over D=1024 ----
    v8f sc = zero;
    for (int d0 = 0; d0 < DH; d0 += 32) {
      Frag16 a, bk;
      const bf16_t* pa = qs + ln * 1032 + d0 + half * 8;
      a.h[0] = *(const v8bf*)pa;
      a.h[1] = *(const v8bf*)(pa + 16);
      const bf16_t* pk = Kb + ((size_t)(b * S_ + kb + ln)) * DH + d0 + half * 8;
      bk.h[0] = *(const v8bf*)pk;
      bk.h[1] = *(const v8bf*)(pk + 16);
      sc = wmma_bf16(a, bk, sc);
    }

    // ---- per-row max within this wave's 16 keys ----
    float rmax[8];
    for (int r = 0; r < 8; ++r) {
      float v = sc[r] * scale;
      sc[r] = v;
      for (int off = 1; off < 16; off <<= 1)
        v = fmaxf(v, __shfl_xor(v, off));
      rmax[r] = v;
    }
    if (ln == 0)
      for (int r = 0; r < 8; ++r) red_m[w][r + half * 8] = rmax[r];
    __syncthreads();  // B1

    // ---- combine max across waves, rescale, exponentiate ----
    float p[8], rsum[8];
    for (int r = 0; r < 8; ++r) {
      const int row = r + half * 8;
      float mn = red_m[0][row];
      for (int ww = 1; ww < 8; ++ww) mn = fmaxf(mn, red_m[ww][row]);
      float m_tot = fmaxf(m_run[r], mn);
      float corr  = __expf(m_run[r] - m_tot);
      m_run[r] = m_tot;
      for (int ni = 0; ni < 8; ++ni) o[ni][r] *= corr;
      float pv = __expf(sc[r] - m_tot);
      p[r] = pv;
      float s = pv;
      for (int off = 1; off < 16; off <<= 1)
        s += __shfl_xor(s, off);
      rsum[r] = s;
      l_run[r] *= corr;
    }
    if (ln == 0)
      for (int r = 0; r < 8; ++r) red_s[w][r + half * 8] = rsum[r];

    // write P (bf16) to LDS: ps[row][w*16 + n]
    for (int r = 0; r < 8; ++r)
      ps[(r + half * 8) * 136 + w * 16 + ln] = (__bf16)p[r];
    __syncthreads();  // B2

    for (int r = 0; r < 8; ++r) {
      const int row = r + half * 8;
      float s = red_s[0][row];
      for (int ww = 1; ww < 8; ++ww) s += red_s[ww][row];
      l_run[r] += s;
    }

    // ---- O += P(16x128) @ V(128 x this wave's 128 cols) ----
    for (int k0 = 0; k0 < 128; k0 += 32) {
      Frag16 a;
      const bf16_t* pa = ps + ln * 136 + k0 + half * 8;
      a.h[0] = *(const v8bf*)pa;
      a.h[1] = *(const v8bf*)(pa + 16);
      for (int ni = 0; ni < 8; ++ni) {
        Frag16 bv;
        const int dcol = w * 128 + ni * 16 + ln;
        const bf16_t* pv = VT + ((size_t)(b * DH + dcol)) * S_ + kc * 128 + k0 + half * 8;
        bv.h[0] = *(const v8bf*)pv;
        bv.h[1] = *(const v8bf*)(pv + 16);
        o[ni] = wmma_bf16(a, bv, o[ni]);
      }
    }
  }

  // ---- epilogue: normalize and store fp32 output ----
  for (int r = 0; r < 8; ++r) {
    const int row = q0 + r + half * 8;
    const float inv = 1.0f / l_run[r];
    for (int ni = 0; ni < 8; ++ni) {
      const int dcol = w * 128 + ni * 16 + ln;
      Out[((size_t)(b * S_ + row)) * DH + dcol] = o[ni][r] * inv;
    }
  }
}

// ---------------------------------------------------------------------------
// Launch. Workspace layout (bytes), total ~134 MB:
//   Xb  @ 0          : 32 MB  (16384x1024 bf16)
//   WTq @ 33554432   : 2 MB   (+WTk, +WTv)
//   Qb  @ 39845888   : 32 MB
//   Kb  @ 73400320   : 32 MB
//   VT  @ 106954752  : 32 MB  ([b][d][s] bf16)
// ---------------------------------------------------------------------------
extern "C" void kernel_launch(void* const* d_in, const int* in_sizes, int n_in,
                              void* d_out, int out_size, void* d_ws, size_t ws_size,
                              hipStream_t stream) {
  const float* x  = (const float*)d_in[0];
  const float* Wq = (const float*)d_in[1];
  const float* bq = (const float*)d_in[2];
  const float* Wk = (const float*)d_in[3];
  const float* bk = (const float*)d_in[4];
  const float* Wv = (const float*)d_in[5];
  const float* bv = (const float*)d_in[6];
  float* out = (float*)d_out;
  char* ws = (char*)d_ws;

  bf16_t* Xb  = (bf16_t*)(ws);
  bf16_t* WTq = (bf16_t*)(ws + 33554432);
  bf16_t* WTk = WTq + (size_t)DIN * DH;
  bf16_t* WTv = WTk + (size_t)DIN * DH;
  bf16_t* Qb  = (bf16_t*)(ws + 39845888);
  bf16_t* Kb  = Qb + (size_t)B_ * S_ * DH;
  bf16_t* VT  = Kb + (size_t)B_ * S_ * DH;

  // x -> bf16 (16M elements as 4M float4)
  k_cvt_x<<<16384, 256, 0, stream>>>(x, Xb, (B_ * S_ * DIN) / 4);
  // W -> bf16 transposed
  k_cvt_w<<<dim3(4096, 3), 256, 0, stream>>>(Wq, Wk, Wv, WTq, WTk, WTv);
  // QKV projections
  dim3 gg(DH / 128, (B_ * S_) / 128);
  k_gemm_qkv<<<gg, 256, 0, stream>>>(Xb, WTq, bq, Qb, 0);
  k_gemm_qkv<<<gg, 256, 0, stream>>>(Xb, WTk, bk, Kb, 0);
  k_gemm_qkv<<<gg, 256, 0, stream>>>(Xb, WTv, bv, VT, 1);
  // Flash attention
  k_attn<<<dim3(S_ / 16, B_), 256, 0, stream>>>(Qb, Kb, VT, out);
}
